// TransformerWithValueHead_1924145348705
// MI455X (gfx1250) — compile-verified, matrix operand order checked by reading, not run on hardware
//
#include <hip/hip_runtime.h>
#include <hip/hip_bf16.h>

#define B_   2
#define S_   1024
#define H_   768
#define L_   12
#define NH_  12
#define V_   32000
#define FF_  3072
#define DH_  64
#define MTOK (B_ * S_)

typedef __attribute__((ext_vector_type(16))) __bf16 bf16x16;
typedef __attribute__((ext_vector_type(8)))  float  f32x8;
typedef __attribute__((ext_vector_type(4)))  unsigned int u32x4;
typedef __attribute__((ext_vector_type(8)))  int          i32x8;
typedef __attribute__((ext_vector_type(4)))  int          i32x4;

__device__ __forceinline__ f32x8 wmma_bf16(bf16x16 a, bf16x16 b, f32x8 c) {
  // D = A(16x32 bf16) * B(32x16 bf16) + C(16x16 f32)
  return __builtin_amdgcn_wmma_f32_16x16x32_bf16(false, a, false, b, (short)0, c,
                                                 false, false);
}

// Fragment load from a bf16 LDS tile stored [row][k] (ISA 7.12.2 layout):
// lanes 0-15 -> row=lane, K-half 0; lanes 16-31 -> row=lane-16, K-half 1.
// VGPR 0..3 hold K = half*8 + {0..7}; VGPR 4..7 hold K = 16 + half*8 + {0..7}.
__device__ __forceinline__ bf16x16 load_frag(const __bf16* base, int ld) {
  const int lane = threadIdx.x & 31;
  const int r    = lane & 15;
  const int half = lane >> 4;
  const __bf16* p = base + r * ld + half * 8;
  bf16x16 v;
#pragma unroll
  for (int i = 0; i < 8; ++i) v[i] = p[i];
#pragma unroll
  for (int i = 0; i < 8; ++i) v[8 + i] = p[16 + i];
  return v;
}

// Same fragment mapping, but source tile is fp32 in LDS (TDM-staged); convert here.
__device__ __forceinline__ bf16x16 load_frag_f32(const float* base, int ld) {
  const int lane = threadIdx.x & 31;
  const int r    = lane & 15;
  const int half = lane >> 4;
  const float* p = base + r * ld + half * 8;
  bf16x16 v;
#pragma unroll
  for (int i = 0; i < 8; ++i) v[i] = (__bf16)p[i];
#pragma unroll
  for (int i = 0; i < 8; ++i) v[8 + i] = (__bf16)p[16 + i];
  return v;
}

// ---------------------------------------------------------------------------
// Tensor Data Mover: DMA a 2D fp32 tile (rows x 32) global -> LDS.
// D# per CDNA5 ISA: group0 {count=1, lds_addr, global_addr,
// type=2}; group1 {data_size=4B, pad 1 DWORD every 32 (row stride 33), tensor
// dims == tile dims (no OOB path), tensor_dim0_stride = K}.
// Issued by wave 0 only; completion via s_wait_tensorcnt + workgroup barrier.
// ---------------------------------------------------------------------------
__device__ __forceinline__ void tdm_load_tile(unsigned lds_off, const float* gaddr,
                                              int rows, int k_stride_elems) {
  const unsigned long long ga = (unsigned long long)(size_t)gaddr;
  u32x4 g0 = {};
  g0[0] = 1u;                                         // count=1, user mode
  g0[1] = lds_off;                                    // LDS byte address
  g0[2] = (unsigned)(ga & 0xFFFFFFFFu);               // global_addr[31:0]
  g0[3] = (unsigned)((ga >> 32) & 0x01FFFFFFu) | (2u << 30);  // ga[56:32] | type=2
  i32x8 g1 = {};
  g1[0] = (2 << 16) | (1 << 20) | (4 << 22);          // 4B elems, pad_en, every 32 DW, +1 DW
  g1[1] = 32 << 16;                                   // tensor_dim0 = 32
  g1[2] = rows << 16;                                 // tensor_dim1 = rows
  g1[3] = 32 << 16;                                   // tile_dim0 = 32
  g1[4] = rows;                                       // tile_dim1 = rows (tile_dim2 = 0)
  g1[5] = k_stride_elems;                             // tensor_dim0_stride (elems)
  i32x4 g2 = {}, g3 = {};
#if defined(__clang_major__) && __clang_major__ >= 23
  i32x8 g4 = {};
  __builtin_amdgcn_tensor_load_to_lds(g0, g1, g2, g3, g4, 0);
#else
  __builtin_amdgcn_tensor_load_to_lds(g0, g1, g2, g3, 0);
#endif
}

__device__ __forceinline__ unsigned lds_offset_of(const void* p) {
  return (unsigned)(size_t)p;  // LDS aperture: low 32 bits = offset in wave's LDS
}

// ---------------------------------------------------------------------------
// Generic GEMM: out[M,N] = act(A[M,K] @ W[N,K]^T + bias[N]) (+ residual)
// Block tile 64(M) x 128(N), K-step 32, 256 threads = 8 waves (2x4),
// each wave computes a 32x32 sub-tile via 4 WMMA accumulators.
// TDM double-buffered staging: wave 0 issues next tile's DMA before computing.
// ---------------------------------------------------------------------------
#define GT_M 64
#define GT_N 128
#define GT_K 32
#define TLD  33   // fp32 units per row in LDS (32 + 1 TDM pad DWORD)

template <bool RELU, bool HASRES>
__global__ __launch_bounds__(256) void gemm_kernel(
    const float* __restrict__ A, const float* __restrict__ W,
    const float* __restrict__ bias, const float* __restrict__ resid,
    float* __restrict__ out, int M, int N, int K, int ldc) {
  __shared__ float As[2][GT_M * TLD];
  __shared__ float Bs[2][GT_N * TLD];

  const int tid  = threadIdx.x;
  const int wid  = tid >> 5;
  const int lane = tid & 31;
  const int m0 = blockIdx.y * GT_M;
  const int n0 = blockIdx.x * GT_N;
  const int wm = wid & 1;   // 0..1 -> 32-row half
  const int wn = wid >> 1;  // 0..3 -> 32-col strip

  f32x8 acc[2][2];
#pragma unroll
  for (int i = 0; i < 2; ++i)
#pragma unroll
    for (int j = 0; j < 2; ++j) acc[i][j] = (f32x8){};

  const float* gA = A + (size_t)m0 * K;  // tile row origin
  const float* gW = W + (size_t)n0 * K;

  // preload K-tile 0 into buffer 0
  if (wid == 0) {
    tdm_load_tile(lds_offset_of(&As[0][0]), gA, GT_M, K);
    tdm_load_tile(lds_offset_of(&Bs[0][0]), gW, GT_N, K);
    __builtin_amdgcn_s_wait_tensorcnt(0);
  }
  __syncthreads();

  int cur = 0;
  for (int kt = 0; kt < K; kt += GT_K) {
    // kick off DMA of the next K-tile into the other buffer (overlaps WMMAs)
    if (kt + GT_K < K && wid == 0) {
      tdm_load_tile(lds_offset_of(&As[cur ^ 1][0]), gA + kt + GT_K, GT_M, K);
      tdm_load_tile(lds_offset_of(&Bs[cur ^ 1][0]), gW + kt + GT_K, GT_N, K);
    }

    bf16x16 af0 = load_frag_f32(&As[cur][(wm * 32 + 0)  * TLD], TLD);
    bf16x16 af1 = load_frag_f32(&As[cur][(wm * 32 + 16) * TLD], TLD);
    bf16x16 bf0 = load_frag_f32(&Bs[cur][(wn * 32 + 0)  * TLD], TLD);
    bf16x16 bf1 = load_frag_f32(&Bs[cur][(wn * 32 + 16) * TLD], TLD);

    acc[0][0] = wmma_bf16(af0, bf0, acc[0][0]);
    acc[0][1] = wmma_bf16(af0, bf1, acc[0][1]);
    acc[1][0] = wmma_bf16(af1, bf0, acc[1][0]);
    acc[1][1] = wmma_bf16(af1, bf1, acc[1][1]);

    if (wid == 0) __builtin_amdgcn_s_wait_tensorcnt(0);
    __syncthreads();
    cur ^= 1;
  }

  // epilogue: C layout -> col = lane&15, row = r + (lane>>4)*8  (branch-free)
  const int cn   = lane & 15;
  const int half = lane >> 4;
#pragma unroll
  for (int j = 0; j < 2; ++j) {
    const int col = n0 + wn * 32 + j * 16 + cn;
    const float bv = bias[col];
#pragma unroll
    for (int i = 0; i < 2; ++i) {
      const int rb = m0 + wm * 32 + i * 16 + half * 8;
      const size_t base = (size_t)rb * ldc + col;
#pragma unroll
      for (int r = 0; r < 8; ++r) {
        float v = acc[i][j][r] + bv;
        if (RELU) v = fmaxf(v, 0.0f);
        const size_t idx = base + (size_t)r * ldc;
        if (HASRES) v += resid[idx];
        out[idx] = v;
      }
    }
  }
}

// ---------------------------------------------------------------------------
// Fused attention (flash-style, online softmax), WMMA for QK^T and P@V.
// qkv: [B*S, 2304] packed q|k|v, feature = h*64+d. out: [B*S, 768].
// Block = 256 threads handles 64 query rows for one (b,h).
// ---------------------------------------------------------------------------
#define QLD  72
#define KLD  72
#define VLD  40
#define PLD  40
#define SLD  36
#define OLD_ 66

__global__ __launch_bounds__(256) void attn_kernel(
    const float* __restrict__ qkv, float* __restrict__ out) {
  __shared__ __bf16 Qs[64 * QLD];   // [qrow][d]
  __shared__ __bf16 Ks[32 * KLD];   // [key][d]
  __shared__ __bf16 Vs[64 * VLD];   // transposed: [d][key]
  __shared__ __bf16 Ps[64 * PLD];   // [qrow][key]  (probs, bf16)
  __shared__ float  Ss[64 * SLD];   // [qrow][key]  scores f32
  __shared__ float  Os[64 * OLD_];  // [qrow][d]    output accumulator
  __shared__ float  mrow[64], lrow[64], frow[64];

  const int tid  = threadIdx.x;
  const int wid  = tid >> 5;
  const int lane = tid & 31;
  const int bh = blockIdx.x;
  const int b  = bh / NH_;
  const int h  = bh % NH_;
  const int q0 = blockIdx.y * 64;
  const size_t baseTok = (size_t)b * S_;

  // load Q tile (64 x 64) once
  {
    const int r = tid >> 2, doff = (tid & 3) * 16;
    const float* g = qkv + (baseTok + q0 + r) * 2304 + h * DH_ + doff;
    __bf16* s = Qs + r * QLD + doff;
#pragma unroll
    for (int i = 0; i < 16; ++i) s[i] = (__bf16)g[i];
  }
  for (int e = tid; e < 64 * 64; e += 256) Os[(e >> 6) * OLD_ + (e & 63)] = 0.0f;
  if (tid < 64) { mrow[tid] = -1e30f; lrow[tid] = 0.0f; }
  __syncthreads();

  for (int k0 = 0; k0 < S_; k0 += 32) {
    // stage K (row-major) and V (transposed)
    {
      const int key = tid >> 3, doff = (tid & 7) * 8;
      const float* gk = qkv + (baseTok + k0 + key) * 2304 + H_ + h * DH_ + doff;
      const float* gv = qkv + (baseTok + k0 + key) * 2304 + 2 * H_ + h * DH_ + doff;
      __bf16* sk = Ks + key * KLD + doff;
#pragma unroll
      for (int i = 0; i < 8; ++i) sk[i] = (__bf16)gk[i];
#pragma unroll
      for (int i = 0; i < 8; ++i) Vs[(doff + i) * VLD + key] = (__bf16)gv[i];
      if (k0 + 32 < S_) {
        __builtin_prefetch(gk + 32 * 2304, 0, 1);
        __builtin_prefetch(gv + 32 * 2304, 0, 1);
      }
    }
    __syncthreads();

    // S = Q @ K^T / 8 : 8 tiles of 16x16, one per wave, K-dim = 64 = 2 WMMAs
    {
      const int tm = wid & 3;   // q 16-row group
      const int tn = wid >> 2;  // key 16-col group (0..1)
      f32x8 sacc = (f32x8){};
      sacc = wmma_bf16(load_frag(Qs + tm * 16 * QLD + 0,  QLD),
                       load_frag(Ks + tn * 16 * KLD + 0,  KLD), sacc);
      sacc = wmma_bf16(load_frag(Qs + tm * 16 * QLD + 32, QLD),
                       load_frag(Ks + tn * 16 * KLD + 32, KLD), sacc);
      const int cn = lane & 15, half = lane >> 4;
      const int col = tn * 16 + cn;
#pragma unroll
      for (int r = 0; r < 8; ++r)
        Ss[(tm * 16 + half * 8 + r) * SLD + col] = sacc[r] * 0.125f;
    }
    __syncthreads();

    // online softmax: one thread per query row
    if (tid < 64) {
      const float mold = mrow[tid];
      float mx = mold;
      const float* srow = Ss + tid * SLD;
      for (int j = 0; j < 32; ++j) mx = fmaxf(mx, srow[j]);
      const float f = __expf(mold - mx);
      float sum = 0.0f;
      __bf16* prow = Ps + tid * PLD;
      for (int j = 0; j < 32; ++j) {
        const float p = __expf(srow[j] - mx);
        prow[j] = (__bf16)p;
        sum += p;
      }
      lrow[tid] = lrow[tid] * f + sum;
      mrow[tid] = mx;
      frow[tid] = f;
    }
    __syncthreads();

    // rescale accumulated O by exp(m_old - m_new)
    for (int e = tid; e < 64 * 64; e += 256) {
      const int r = e >> 6, c = e & 63;
      Os[r * OLD_ + c] *= frow[r];
    }
    __syncthreads();

    // O += P @ V : 16 tiles of 16x16, two per wave, K-dim = 32 = 1 WMMA
#pragma unroll
    for (int tt = 0; tt < 2; ++tt) {
      const int t  = wid * 2 + tt;
      const int tm = t >> 2;   // q 16-row group (0..3)
      const int tn = t & 3;    // d 16-col group (0..3)
      const int cn = lane & 15, half = lane >> 4;
      const int col = tn * 16 + cn;
      f32x8 oacc;
#pragma unroll
      for (int r = 0; r < 8; ++r) oacc[r] = Os[(tm * 16 + half * 8 + r) * OLD_ + col];
      oacc = wmma_bf16(load_frag(Ps + tm * 16 * PLD, PLD),
                       load_frag(Vs + tn * 16 * VLD, VLD), oacc);
#pragma unroll
      for (int r = 0; r < 8; ++r) Os[(tm * 16 + half * 8 + r) * OLD_ + col] = oacc[r];
    }
    __syncthreads();
  }

  // normalize and write out (concat heads layout)
  for (int e = tid; e < 64 * 64; e += 256) {
    const int r = e >> 6, c = e & 63;
    out[(baseTok + q0 + r) * H_ + h * DH_ + c] = Os[r * OLD_ + c] / lrow[r];
  }
}

// ---------------------------------------------------------------------------
// LayerNorm over H=768: one block per row.
// ---------------------------------------------------------------------------
__global__ __launch_bounds__(256) void ln_kernel(
    const float* __restrict__ x, float* __restrict__ out,
    const float* __restrict__ g, const float* __restrict__ bta) {
  __shared__ float red[256];
  const int row = blockIdx.x;
  const int tid = threadIdx.x;
  const float* xr = x + (size_t)row * H_;
  float v[3];
  float s = 0.0f;
#pragma unroll
  for (int i = 0; i < 3; ++i) { v[i] = xr[tid + i * 256]; s += v[i]; }
  red[tid] = s;
  __syncthreads();
  for (int o = 128; o > 0; o >>= 1) {
    if (tid < o) red[tid] += red[tid + o];
    __syncthreads();
  }
  const float mean = red[0] * (1.0f / H_);
  __syncthreads();
  float s2 = 0.0f;
#pragma unroll
  for (int i = 0; i < 3; ++i) { const float d = v[i] - mean; s2 += d * d; }
  red[tid] = s2;
  __syncthreads();
  for (int o = 128; o > 0; o >>= 1) {
    if (tid < o) red[tid] += red[tid + o];
    __syncthreads();
  }
  const float inv = rsqrtf(red[0] * (1.0f / H_) + 1e-5f);
  float* orow = out + (size_t)row * H_;
#pragma unroll
  for (int i = 0; i < 3; ++i) {
    const int c = tid + i * 256;
    orow[c] = (v[i] - mean) * inv * g[c] + bta[c];
  }
}

// ---------------------------------------------------------------------------
// Step-position cumsum (per batch row, serial — tiny) and embedding gather.
// ---------------------------------------------------------------------------
__global__ void steps_kernel(const int* __restrict__ ids, int* __restrict__ steps) {
  const int b = blockIdx.x;
  if (threadIdx.x == 0) {
    int cnt = 0;
    for (int s = 0; s < S_; ++s) {
      const int id = ids[b * S_ + s];
      if (id == 1) {
        ++cnt;
        steps[b * S_ + s] = cnt > 9 ? 9 : cnt;  // clamp like jnp OOB gather
      } else {
        steps[b * S_ + s] = 0;
      }
    }
  }
}

__global__ __launch_bounds__(256) void embed_kernel(
    const int* __restrict__ ids, const int* __restrict__ steps,
    const float* __restrict__ tok, const float* __restrict__ pos,
    const float* __restrict__ rea, const float* __restrict__ stp,
    float* __restrict__ out) {
  const int t  = blockIdx.x;       // token index 0..MTOK-1
  const int s  = t % S_;
  const int id = ids[t];
  const int rid = (id == 1) ? 1 : (id == 2) ? 2 : (id == 3) ? 3 : 0;
  const int sp  = steps[t];
  const float* te = tok + (size_t)id * H_;
  const float* pe = pos + (size_t)s * H_;
  const float* re = rea + (size_t)rid * H_;
  const float* se = stp + (size_t)sp * H_;
  float* o = out + (size_t)t * H_;
#pragma unroll
  for (int i = 0; i < 3; ++i) {
    const int c = threadIdx.x + i * 256;
    o[c] = te[c] + pe[c] + re[c] + se[c];
  }
}

// ---------------------------------------------------------------------------
// Host-side orchestration
// ---------------------------------------------------------------------------
static inline void launch_gemm(hipStream_t st, const float* A, const float* W,
                               const float* bias, const float* resid, float* out,
                               int M, int N, int K, int ldc, int relu) {
  dim3 grid(N / GT_N, M / GT_M);
  if (resid)
    gemm_kernel<false, true><<<grid, 256, 0, st>>>(A, W, bias, resid, out, M, N, K, ldc);
  else if (relu)
    gemm_kernel<true, false><<<grid, 256, 0, st>>>(A, W, bias, nullptr, out, M, N, K, ldc);
  else
    gemm_kernel<false, false><<<grid, 256, 0, st>>>(A, W, bias, nullptr, out, M, N, K, ldc);
}

extern "C" void kernel_launch(void* const* d_in, const int* in_sizes, int n_in,
                              void* d_out, int out_size, void* d_ws, size_t ws_size,
                              hipStream_t stream) {
  (void)in_sizes; (void)n_in; (void)out_size; (void)ws_size;

  const int*   input_ids  = (const int*)d_in[0];
  const float* tok_emb    = (const float*)d_in[1];
  const float* pos_emb    = (const float*)d_in[2];
  const float* reason_emb = (const float*)d_in[3];
  const float* step_emb   = (const float*)d_in[4];
  const float* emb_ln_g   = (const float*)d_in[5];
  const float* emb_ln_b   = (const float*)d_in[6];
  const float* enc_qkv_w  = (const float*)d_in[7];
  const float* enc_qkv_b  = (const float*)d_in[8];
  const float* enc_out_w  = (const float*)d_in[9];
  const float* enc_out_b  = (const float*)d_in[10];
  const float* enc_ln1_g  = (const float*)d_in[11];
  const float* enc_ln1_b  = (const float*)d_in[12];
  const float* enc_ff1_w  = (const float*)d_in[13];
  const float* enc_ff1_b  = (const float*)d_in[14];
  const float* enc_ff2_w  = (const float*)d_in[15];
  const float* enc_ff2_b  = (const float*)d_in[16];
  const float* enc_ln2_g  = (const float*)d_in[17];
  const float* enc_ln2_b  = (const float*)d_in[18];
  const float* dec_sqkv_w = (const float*)d_in[19];
  const float* dec_sqkv_b = (const float*)d_in[20];
  const float* dec_sout_w = (const float*)d_in[21];
  const float* dec_sout_b = (const float*)d_in[22];
  const float* dec_ln1_g  = (const float*)d_in[23];
  const float* dec_ln1_b  = (const float*)d_in[24];
  const float* dec_cqkv_w = (const float*)d_in[25];
  const float* dec_cqkv_b = (const float*)d_in[26];
  const float* dec_cout_w = (const float*)d_in[27];
  const float* dec_cout_b = (const float*)d_in[28];
  const float* dec_ln2_g  = (const float*)d_in[29];
  const float* dec_ln2_b  = (const float*)d_in[30];
  const float* dec_ff1_w  = (const float*)d_in[31];
  const float* dec_ff1_b  = (const float*)d_in[32];
  const float* dec_ff2_w  = (const float*)d_in[33];
  const float* dec_ff2_b  = (const float*)d_in[34];
  const float* dec_ln3_g  = (const float*)d_in[35];
  const float* dec_ln3_b  = (const float*)d_in[36];
  const float* lm_w       = (const float*)d_in[37];
  const float* lm_b       = (const float*)d_in[38];

  // workspace layout
  float* emb  = (float*)d_ws;                       // [MTOK, H]  (kept for cross-attn)
  float* x    = emb  + (size_t)MTOK * H_;           // [MTOK, H]  running stream
  float* attn = x    + (size_t)MTOK * H_;           // [MTOK, H]
  float* qkv  = attn + (size_t)MTOK * H_;           // [MTOK, 3H]
  float* ffh  = qkv  + (size_t)MTOK * 3 * H_;       // [MTOK, FF]
  int*   stpp = (int*)(ffh + (size_t)MTOK * FF_);   // [MTOK]

  const dim3 attnGrid(B_ * NH_, S_ / 64);

  steps_kernel<<<B_, 32, 0, stream>>>(input_ids, stpp);
  embed_kernel<<<MTOK, 256, 0, stream>>>(input_ids, stpp, tok_emb, pos_emb,
                                         reason_emb, step_emb, emb);
  ln_kernel<<<MTOK, 256, 0, stream>>>(emb, emb, emb_ln_g, emb_ln_b);
  hipMemcpyAsync(x, emb, (size_t)MTOK * H_ * sizeof(float),
                 hipMemcpyDeviceToDevice, stream);

  const size_t wQKV = (size_t)3 * H_ * H_;
  const size_t wOUT = (size_t)H_ * H_;
  const size_t wFF  = (size_t)FF_ * H_;

  // ---------------- encoder ----------------
  for (int l = 0; l < L_; ++l) {
    launch_gemm(stream, x, enc_qkv_w + l * wQKV, enc_qkv_b + (size_t)l * 3 * H_,
                nullptr, qkv, MTOK, 3 * H_, H_, 3 * H_, 0);
    attn_kernel<<<attnGrid, 256, 0, stream>>>(qkv, attn);
    launch_gemm(stream, attn, enc_out_w + l * wOUT, enc_out_b + (size_t)l * H_,
                x, x, MTOK, H_, H_, H_, 0);
    ln_kernel<<<MTOK, 256, 0, stream>>>(x, x, enc_ln1_g + (size_t)l * H_,
                                        enc_ln1_b + (size_t)l * H_);
    launch_gemm(stream, x, enc_ff1_w + l * wFF, enc_ff1_b + (size_t)l * FF_,
                nullptr, ffh, MTOK, FF_, H_, FF_, 1);
    launch_gemm(stream, ffh, enc_ff2_w + l * wFF, enc_ff2_b + (size_t)l * H_,
                x, x, MTOK, H_, FF_, H_, 0);
    ln_kernel<<<MTOK, 256, 0, stream>>>(x, x, enc_ln2_g + (size_t)l * H_,
                                        enc_ln2_b + (size_t)l * H_);
  }

  // ---------------- decoder (tgt = encoder output, memory = emb) ----------------
  for (int l = 0; l < L_; ++l) {
    // self-attention
    launch_gemm(stream, x, dec_sqkv_w + l * wQKV, dec_sqkv_b + (size_t)l * 3 * H_,
                nullptr, qkv, MTOK, 3 * H_, H_, 3 * H_, 0);
    attn_kernel<<<attnGrid, 256, 0, stream>>>(qkv, attn);
    launch_gemm(stream, attn, dec_sout_w + l * wOUT, dec_sout_b + (size_t)l * H_,
                x, x, MTOK, H_, H_, H_, 0);
    ln_kernel<<<MTOK, 256, 0, stream>>>(x, x, dec_ln1_g + (size_t)l * H_,
                                        dec_ln1_b + (size_t)l * H_);
    // cross-attention: q from x, k|v from emb, packed into qkv[:, 0:768 | 768:2304]
    launch_gemm(stream, x, dec_cqkv_w + l * wQKV, dec_cqkv_b + (size_t)l * 3 * H_,
                nullptr, qkv, MTOK, H_, H_, 3 * H_, 0);
    launch_gemm(stream, emb, dec_cqkv_w + l * wQKV + (size_t)H_ * H_,
                dec_cqkv_b + (size_t)l * 3 * H_ + H_,
                nullptr, qkv + H_, MTOK, 2 * H_, H_, 3 * H_, 0);
    attn_kernel<<<attnGrid, 256, 0, stream>>>(qkv, attn);
    launch_gemm(stream, attn, dec_cout_w + l * wOUT, dec_cout_b + (size_t)l * H_,
                x, x, MTOK, H_, H_, H_, 0);
    ln_kernel<<<MTOK, 256, 0, stream>>>(x, x, dec_ln2_g + (size_t)l * H_,
                                        dec_ln2_b + (size_t)l * H_);
    // FFN
    launch_gemm(stream, x, dec_ff1_w + l * wFF, dec_ff1_b + (size_t)l * FF_,
                nullptr, ffh, MTOK, FF_, H_, FF_, 1);
    launch_gemm(stream, ffh, dec_ff2_w + l * wFF, dec_ff2_b + (size_t)l * H_,
                x, x, MTOK, H_, FF_, H_, 0);
    ln_kernel<<<MTOK, 256, 0, stream>>>(x, x, dec_ln3_g + (size_t)l * H_,
                                        dec_ln3_b + (size_t)l * H_);
  }

  // ---------------- LM head ----------------
  launch_gemm(stream, x, lm_w, lm_b, nullptr, (float*)d_out, MTOK, V_, H_, V_, 0);
}
